// LocalStream_1443109011696
// MI455X (gfx1250) — matrix operand order repaired and checked by reading, not compiled, not run on hardware
//
#include <hip/hip_runtime.h>

// ---------------------------------------------------------------------------
// Types / WMMA helpers (CDNA5 gfx1250, wave32)
// ---------------------------------------------------------------------------
typedef _Float16 half_t;
typedef __attribute__((ext_vector_type(16))) _Float16 v16h;
typedef __attribute__((ext_vector_type(8)))  _Float16 v8h;
typedef __attribute__((ext_vector_type(8)))  float    v8f;

#define WPB  4  // waves per block (node kernels)
#define EWPB 2  // waves per block (edge kernel, M=32 per wave)

__device__ __forceinline__ v8f wmma16(v16h a, v16h b, v8f c) {
  return __builtin_amdgcn_wmma_f32_16x16x32_f16(false, a, false, b, (short)0, c,
                                                false, false);
}

__device__ __forceinline__ v8f zero8() {
  v8f r;
#pragma unroll
  for (int i = 0; i < 8; ++i) r[i] = 0.f;
  return r;
}

// A-fragment (16x32, 16-bit) per ISA layout:
//   lane L: row M = L&15 ; half i<8 -> K = (L>>4)*8 + i ; half i>=8 -> +16
__device__ __forceinline__ v16h frag_a_f16(const half_t* rowBase, int colBase, int kg) {
  v8h lo = *(const v8h*)(rowBase + colBase + kg * 8);
  v8h hi = *(const v8h*)(rowBase + colBase + kg * 8 + 16);
  v16h r;
#pragma unroll
  for (int i = 0; i < 8; ++i) { r[i] = lo[i]; r[i + 8] = hi[i]; }
  return r;
}

__device__ __forceinline__ v16h frag_a_f32(const float* rowBase, int colBase, int kg) {
  v16h r;
#pragma unroll
  for (int i = 0; i < 8; ++i) {
    r[i]     = (half_t)rowBase[colBase + kg * 8 + i];
    r[i + 8] = (half_t)rowBase[colBase + kg * 8 + 16 + i];
  }
  return r;
}

// B-fragment: pre-packed so each lane loads 16 contiguous halves (32 B)
__device__ __forceinline__ v16h frag_b(const half_t* __restrict__ packed, int frag, int lane) {
  return *(const v16h*)(packed + ((size_t)frag * 32 + lane) * 16);
}

#define WAIT_DS() asm volatile("s_wait_dscnt 0x0" ::: "memory")

// ---------------------------------------------------------------------------
// Weight packing: row-major f32 [K x N] (or transposed [N x K]) ->
// fragment-packed f16, ISA B layout: lane L: col N = L&15, half i<8 -> K=(L>>4)*8+i ; half i>=8 -> +16
// ---------------------------------------------------------------------------
__global__ void pack_w(const float* __restrict__ W, half_t* __restrict__ out,
                       int K, int Ncols, int transposed) {
  int idx = blockIdx.x * blockDim.x + threadIdx.x;
  if (idx >= K * Ncols) return;
  int i    = idx & 15;
  int lane = (idx >> 4) & 31;
  int frag = idx >> 9;
  int ntiles = Ncols >> 4;
  int nt = frag % ntiles;
  int kc = frag / ntiles;
  int col = nt * 16 + (lane & 15);
  int kr  = kc * 32 + (lane >> 4) * 8 + (i & 7) + (i >> 3) * 16;
  float v = transposed ? W[(size_t)col * K + kr] : W[(size_t)kr * Ncols + col];
  out[idx] = (half_t)v;
}

// ---------------------------------------------------------------------------
// Encoder: h = relu(x @ W1 + b1) @ W2 + b2  (128->128->128)
// ---------------------------------------------------------------------------
__global__ void enc_kernel(const float* __restrict__ x,
                           const half_t* __restrict__ w1p, const float* __restrict__ b1,
                           const half_t* __restrict__ w2p, const float* __restrict__ b2,
                           float* __restrict__ h, half_t* __restrict__ h16, int nTiles) {
  __shared__ half_t st[WPB][16 * 128];
  const int lane = threadIdx.x & 31, wv = threadIdx.x >> 5;
  const int m = lane & 15, kg = lane >> 4;
  const int tile = blockIdx.x * WPB + wv;
  if (tile >= nTiles) return;
  const float* xr = x + (size_t)(tile * 16 + m) * 128;

  v16h a[4];
#pragma unroll
  for (int kc = 0; kc < 4; ++kc) a[kc] = frag_a_f32(xr, kc * 32, kg);

#pragma unroll 1
  for (int nt = 0; nt < 8; ++nt) {
    float bias = b1[nt * 16 + m];
    v8f acc = zero8();
#pragma unroll
    for (int kc = 0; kc < 4; ++kc)
      acc = wmma16(a[kc], frag_b(w1p, kc * 8 + nt, lane), acc);
#pragma unroll
    for (int r = 0; r < 8; ++r)
      st[wv][(8 * kg + r) * 128 + nt * 16 + m] = (half_t)fmaxf(acc[r] + bias, 0.f);
  }
  WAIT_DS();
#pragma unroll
  for (int kc = 0; kc < 4; ++kc) a[kc] = frag_a_f16(&st[wv][m * 128], kc * 32, kg);

#pragma unroll 1
  for (int nt = 0; nt < 8; ++nt) {
    float bias = b2[nt * 16 + m];
    v8f acc = zero8();
#pragma unroll
    for (int kc = 0; kc < 4; ++kc)
      acc = wmma16(a[kc], frag_b(w2p, kc * 8 + nt, lane), acc);
#pragma unroll
    for (int r = 0; r < 8; ++r) {
      int grow = tile * 16 + 8 * kg + r;
      int col  = nt * 16 + m;
      float o  = acc[r] + bias;
      h[(size_t)grow * 128 + col]   = o;
      h16[(size_t)grow * 128 + col] = (half_t)o;
    }
  }
}

// ---------------------------------------------------------------------------
// Edge pipeline: one wave handles 32 edges; accumulator-resident form.
// All loops touching acc0[]/acc1[] are FULLY unrolled so the 128-VGPR
// accumulator state never leaves registers (runtime indexing => scratch).
// ---------------------------------------------------------------------------
__global__ void edge_kernel(const half_t* __restrict__ h16,
                            const float* __restrict__ edge_attr,
                            const int* __restrict__ srcI, const int* __restrict__ dstI,
                            const half_t* __restrict__ w1p, const float* __restrict__ b1,
                            const half_t* __restrict__ w2p, const float* __restrict__ b2,
                            const half_t* __restrict__ m1p, const float* __restrict__ mb1,
                            const half_t* __restrict__ m2p, const float* __restrict__ mb2,
                            float* __restrict__ aggr, int nTiles) {
  __shared__ half_t stA[EWPB][32 * 128];
  __shared__ half_t stB[EWPB][32 * 128];
  __shared__ int    dI[EWPB][32];
  const int lane = threadIdx.x & 31, wv = threadIdx.x >> 5;
  const int m = lane & 15, kg = lane >> 4;
  const int tile = blockIdx.x * EWPB + wv;
  if (tile >= nTiles) return;

  const int e0 = tile * 32 + m;
  const int e1 = e0 + 16;
  const int s0 = srcI[e0], d0 = dstI[e0];
  const int s1 = srcI[e1], d1 = dstI[e1];
  if (kg == 0) { dI[wv][m] = d0; dI[wv][16 + m] = d1; }
  const half_t* hd0 = h16 + (size_t)d0 * 128;
  const half_t* hd1 = h16 + (size_t)d1 * 128;
  const half_t* hs0 = h16 + (size_t)s0 * 128;
  const half_t* hs1 = h16 + (size_t)s1 * 128;

  v8f acc0[8], acc1[8];

  // ---------------- stage 1: 288 -> 128, relu -> stA ----------------
#pragma unroll
  for (int nt = 0; nt < 8; ++nt) { acc0[nt] = zero8(); acc1[nt] = zero8(); }
#pragma unroll 1
  for (int kc = 0; kc < 4; ++kc) {            // dst-feature chunks
    v16h x0 = frag_a_f16(hd0, kc * 32, kg);
    v16h x1 = frag_a_f16(hd1, kc * 32, kg);
#pragma unroll
    for (int nt = 0; nt < 8; ++nt) {
      v16h b = frag_b(w1p, kc * 8 + nt, lane);
      acc0[nt] = wmma16(x0, b, acc0[nt]);
      acc1[nt] = wmma16(x1, b, acc1[nt]);
    }
  }
#pragma unroll 1
  for (int kc = 0; kc < 4; ++kc) {            // src-feature chunks
    v16h x0 = frag_a_f16(hs0, kc * 32, kg);
    v16h x1 = frag_a_f16(hs1, kc * 32, kg);
#pragma unroll
    for (int nt = 0; nt < 8; ++nt) {
      v16h b = frag_b(w1p, (4 + kc) * 8 + nt, lane);
      acc0[nt] = wmma16(x0, b, acc0[nt]);
      acc1[nt] = wmma16(x1, b, acc1[nt]);
    }
  }
  {                                           // edge_attr chunk
    v16h x0 = frag_a_f32(edge_attr + (size_t)e0 * 32, 0, kg);
    v16h x1 = frag_a_f32(edge_attr + (size_t)e1 * 32, 0, kg);
#pragma unroll
    for (int nt = 0; nt < 8; ++nt) {
      v16h b = frag_b(w1p, 64 + nt, lane);
      acc0[nt] = wmma16(x0, b, acc0[nt]);
      acc1[nt] = wmma16(x1, b, acc1[nt]);
    }
  }
#pragma unroll
  for (int nt = 0; nt < 8; ++nt) {
    float bias = b1[nt * 16 + m];
#pragma unroll
    for (int r = 0; r < 8; ++r) {
      stA[wv][(8 * kg + r) * 128 + nt * 16 + m]      = (half_t)fmaxf(acc0[nt][r] + bias, 0.f);
      stA[wv][(16 + 8 * kg + r) * 128 + nt * 16 + m] = (half_t)fmaxf(acc1[nt][r] + bias, 0.f);
    }
  }
  WAIT_DS();

  // ---------------- stage 2: 128 -> 128 (+b2) -> stB ----------------
#pragma unroll
  for (int nt = 0; nt < 8; ++nt) { acc0[nt] = zero8(); acc1[nt] = zero8(); }
#pragma unroll 1
  for (int kc = 0; kc < 4; ++kc) {
    v16h x0 = frag_a_f16(&stA[wv][m * 128], kc * 32, kg);
    v16h x1 = frag_a_f16(&stA[wv][(16 + m) * 128], kc * 32, kg);
#pragma unroll
    for (int nt = 0; nt < 8; ++nt) {
      v16h b = frag_b(w2p, kc * 8 + nt, lane);
      acc0[nt] = wmma16(x0, b, acc0[nt]);
      acc1[nt] = wmma16(x1, b, acc1[nt]);
    }
  }
#pragma unroll
  for (int nt = 0; nt < 8; ++nt) {
    float bias = b2[nt * 16 + m];
#pragma unroll
    for (int r = 0; r < 8; ++r) {
      stB[wv][(8 * kg + r) * 128 + nt * 16 + m]      = (half_t)(acc0[nt][r] + bias);
      stB[wv][(16 + 8 * kg + r) * 128 + nt * 16 + m] = (half_t)(acc1[nt][r] + bias);
    }
  }
  WAIT_DS();

  // ---------------- stage 3: 128 -> 128, relu -> stA ----------------
#pragma unroll
  for (int nt = 0; nt < 8; ++nt) { acc0[nt] = zero8(); acc1[nt] = zero8(); }
#pragma unroll 1
  for (int kc = 0; kc < 4; ++kc) {
    v16h x0 = frag_a_f16(&stB[wv][m * 128], kc * 32, kg);
    v16h x1 = frag_a_f16(&stB[wv][(16 + m) * 128], kc * 32, kg);
#pragma unroll
    for (int nt = 0; nt < 8; ++nt) {
      v16h b = frag_b(m1p, kc * 8 + nt, lane);
      acc0[nt] = wmma16(x0, b, acc0[nt]);
      acc1[nt] = wmma16(x1, b, acc1[nt]);
    }
  }
#pragma unroll
  for (int nt = 0; nt < 8; ++nt) {
    float bias = mb1[nt * 16 + m];
#pragma unroll
    for (int r = 0; r < 8; ++r) {
      stA[wv][(8 * kg + r) * 128 + nt * 16 + m]      = (half_t)fmaxf(acc0[nt][r] + bias, 0.f);
      stA[wv][(16 + 8 * kg + r) * 128 + nt * 16 + m] = (half_t)fmaxf(acc1[nt][r] + bias, 0.f);
    }
  }
  WAIT_DS();

  // ---------------- stage 4: 128 -> 128 (+mb2) -> atomic scatter ----
#pragma unroll
  for (int nt = 0; nt < 8; ++nt) { acc0[nt] = zero8(); acc1[nt] = zero8(); }
#pragma unroll 1
  for (int kc = 0; kc < 4; ++kc) {
    v16h x0 = frag_a_f16(&stA[wv][m * 128], kc * 32, kg);
    v16h x1 = frag_a_f16(&stA[wv][(16 + m) * 128], kc * 32, kg);
#pragma unroll
    for (int nt = 0; nt < 8; ++nt) {
      v16h b = frag_b(m2p, kc * 8 + nt, lane);
      acc0[nt] = wmma16(x0, b, acc0[nt]);
      acc1[nt] = wmma16(x1, b, acc1[nt]);
    }
  }
#pragma unroll
  for (int nt = 0; nt < 8; ++nt) {
    float bias = mb2[nt * 16 + m];
#pragma unroll
    for (int r = 0; r < 8; ++r) {
      atomicAdd(&aggr[(size_t)dI[wv][8 * kg + r] * 128 + nt * 16 + m],      acc0[nt][r] + bias);
      atomicAdd(&aggr[(size_t)dI[wv][16 + 8 * kg + r] * 128 + nt * 16 + m], acc1[nt][r] + bias);
    }
  }
}

// ---------------------------------------------------------------------------
// Node update: preln = h + MLP2([h | aggr], W1(256x128), W2(128x128))
// ---------------------------------------------------------------------------
__global__ void upd_kernel(const half_t* __restrict__ h16, const float* __restrict__ aggr,
                           const float* __restrict__ h,
                           const half_t* __restrict__ w1p, const float* __restrict__ b1,
                           const half_t* __restrict__ w2p, const float* __restrict__ b2,
                           float* __restrict__ preln, int nTiles) {
  __shared__ half_t st[WPB][16 * 128];
  const int lane = threadIdx.x & 31, wv = threadIdx.x >> 5;
  const int m = lane & 15, kg = lane >> 4;
  const int tile = blockIdx.x * WPB + wv;
  if (tile >= nTiles) return;
  const int node = tile * 16 + m;

  v16h a[8];
#pragma unroll
  for (int kc = 0; kc < 4; ++kc) {
    a[kc]     = frag_a_f16(h16 + (size_t)node * 128, kc * 32, kg);
    a[4 + kc] = frag_a_f32(aggr + (size_t)node * 128, kc * 32, kg);
  }
#pragma unroll 1
  for (int nt = 0; nt < 8; ++nt) {
    float bias = b1[nt * 16 + m];
    v8f acc = zero8();
#pragma unroll
    for (int kc = 0; kc < 8; ++kc)
      acc = wmma16(a[kc], frag_b(w1p, kc * 8 + nt, lane), acc);
#pragma unroll
    for (int r = 0; r < 8; ++r)
      st[wv][(8 * kg + r) * 128 + nt * 16 + m] = (half_t)fmaxf(acc[r] + bias, 0.f);
  }
  WAIT_DS();
#pragma unroll
  for (int kc = 0; kc < 4; ++kc) a[kc] = frag_a_f16(&st[wv][m * 128], kc * 32, kg);

#pragma unroll 1
  for (int nt = 0; nt < 8; ++nt) {
    float bias = b2[nt * 16 + m];
    v8f acc = zero8();
#pragma unroll
    for (int kc = 0; kc < 4; ++kc)
      acc = wmma16(a[kc], frag_b(w2p, kc * 8 + nt, lane), acc);
#pragma unroll
    for (int r = 0; r < 8; ++r) {
      int grow = tile * 16 + 8 * kg + r;
      int col  = nt * 16 + m;
      preln[(size_t)grow * 128 + col] = acc[r] + bias + h[(size_t)grow * 128 + col];
    }
  }
}

// ---------------------------------------------------------------------------
// LayerNorm over 128 cols, one wave32 per row; writes h (f32) + h16 (f16)
// ---------------------------------------------------------------------------
__global__ void ln_kernel(const float* __restrict__ pre, const float* __restrict__ g,
                          const float* __restrict__ b, float* __restrict__ h,
                          half_t* __restrict__ h16, int Nrows) {
  int wid  = (blockIdx.x * blockDim.x + threadIdx.x) >> 5;
  int lane = threadIdx.x & 31;
  if (wid >= Nrows) return;
  const float* row = pre + (size_t)wid * 128;
  float v[4], s = 0.f, sq = 0.f;
#pragma unroll
  for (int j = 0; j < 4; ++j) { v[j] = row[lane + 32 * j]; s += v[j]; sq += v[j] * v[j]; }
#pragma unroll
  for (int o = 16; o > 0; o >>= 1) { s += __shfl_xor(s, o, 32); sq += __shfl_xor(sq, o, 32); }
  float mean = s * (1.f / 128.f);
  float var  = sq * (1.f / 128.f) - mean * mean;
  float rstd = rsqrtf(var + 1e-5f);
#pragma unroll
  for (int j = 0; j < 4; ++j) {
    int c = lane + 32 * j;
    float o = (v[j] - mean) * rstd * g[c] + b[c];
    h[(size_t)wid * 128 + c]   = o;
    h16[(size_t)wid * 128 + c] = (half_t)o;
  }
}

// ---------------------------------------------------------------------------
// GRU cell: per wave, 16 nodes x 16 cols; six accumulators in registers
// ---------------------------------------------------------------------------
__global__ void gru_kernel(const half_t* __restrict__ h16, const float* __restrict__ h_prev,
                           const half_t* __restrict__ wihp, const half_t* __restrict__ whhp,
                           const float* __restrict__ bih, const float* __restrict__ bhh,
                           float* __restrict__ out, int nTiles) {
  const int lane = threadIdx.x & 31, wv = threadIdx.x >> 5;
  const int task = blockIdx.x * WPB + wv;
  if (task >= nTiles * 8) return;
  const int tile = task >> 3;
  const int t    = task & 7;  // column-group: cols [16t, 16t+16)
  const int m = lane & 15, kg = lane >> 4;
  const int node = tile * 16 + m;
  const int c = t * 16 + m;

  v16h aI[4], aH[4];
#pragma unroll
  for (int kc = 0; kc < 4; ++kc) {
    aI[kc] = frag_a_f16(h16 + (size_t)node * 128, kc * 32, kg);
    aH[kc] = frag_a_f32(h_prev + (size_t)node * 128, kc * 32, kg);
  }
  v8f air = zero8(), aiz = zero8(), ain = zero8();
  v8f ahr = zero8(), ahz = zero8(), ahn = zero8();
#pragma unroll
  for (int kc = 0; kc < 4; ++kc) {
    air = wmma16(aI[kc], frag_b(wihp, kc * 24 + t,      lane), air);
    aiz = wmma16(aI[kc], frag_b(wihp, kc * 24 + 8 + t,  lane), aiz);
    ain = wmma16(aI[kc], frag_b(wihp, kc * 24 + 16 + t, lane), ain);
    ahr = wmma16(aH[kc], frag_b(whhp, kc * 24 + t,      lane), ahr);
    ahz = wmma16(aH[kc], frag_b(whhp, kc * 24 + 8 + t,  lane), ahz);
    ahn = wmma16(aH[kc], frag_b(whhp, kc * 24 + 16 + t, lane), ahn);
  }
  float bir = bih[c], biz = bih[128 + c], bin = bih[256 + c];
  float bhr = bhh[c], bhz = bhh[128 + c], bhn = bhh[256 + c];
#pragma unroll
  for (int r = 0; r < 8; ++r) {
    int grow = tile * 16 + 8 * kg + r;
    float hp = h_prev[(size_t)grow * 128 + c];
    float rg = 1.f / (1.f + __expf(-(air[r] + bir + ahr[r] + bhr)));
    float z  = 1.f / (1.f + __expf(-(aiz[r] + biz + ahz[r] + bhz)));
    float nn = tanhf(ain[r] + bin + rg * (ahn[r] + bhn));
    out[(size_t)grow * 128 + c] = (1.f - z) * nn + z * hp;
  }
}

// ---------------------------------------------------------------------------
// Host orchestration
// ---------------------------------------------------------------------------
extern "C" void kernel_launch(void* const* d_in, const int* in_sizes, int n_in,
                              void* d_out, int out_size, void* d_ws, size_t ws_size,
                              hipStream_t stream) {
  (void)in_sizes; (void)n_in; (void)out_size; (void)ws_size;
  constexpr int Nn = 50000, Ee = 800000;

  const float* x        = (const float*)d_in[0];
  const float* edge_attr= (const float*)d_in[1];
  const float* h_prev   = (const float*)d_in[2];
  const float* enc_w1   = (const float*)d_in[3];
  const float* enc_b1   = (const float*)d_in[4];
  const float* enc_w2   = (const float*)d_in[5];
  const float* enc_b2   = (const float*)d_in[6];
  const float* et_w1    = (const float*)d_in[7];
  const float* et_b1    = (const float*)d_in[8];
  const float* et_w2    = (const float*)d_in[9];
  const float* et_b2    = (const float*)d_in[10];
  const float* msg_w1   = (const float*)d_in[11];
  const float* msg_b1   = (const float*)d_in[12];
  const float* msg_w2   = (const float*)d_in[13];
  const float* msg_b2   = (const float*)d_in[14];
  const float* upd_w1   = (const float*)d_in[15];
  const float* upd_b1   = (const float*)d_in[16];
  const float* upd_w2   = (const float*)d_in[17];
  const float* upd_b2   = (const float*)d_in[18];
  const float* ln_g     = (const float*)d_in[19];
  const float* ln_b     = (const float*)d_in[20];
  const float* gru_wih  = (const float*)d_in[21];
  const float* gru_whh  = (const float*)d_in[22];
  const float* gru_bih  = (const float*)d_in[23];
  const float* gru_bhh  = (const float*)d_in[24];
  const int*   eidx     = (const int*)d_in[25];
  const int*   srcI = eidx;
  const int*   dstI = eidx + Ee;

  char* base = (char*)d_ws;
  size_t off = 0;
  auto alloc = [&](size_t bytes) -> void* {
    void* p = base + off;
    off = (off + bytes + 255) & ~(size_t)255;
    return p;
  };
  float*  h     = (float*)alloc((size_t)Nn * 128 * 4);
  half_t* h16   = (half_t*)alloc((size_t)Nn * 128 * 2);
  float*  aggr  = (float*)alloc((size_t)Nn * 128 * 4);
  float*  preln = (float*)alloc((size_t)Nn * 128 * 4);
  half_t* enc1p = (half_t*)alloc((size_t)16384 * 2);
  half_t* enc2p = (half_t*)alloc((size_t)16384 * 2);
  half_t* et1p  = (half_t*)alloc((size_t)4 * 36864 * 2);
  half_t* et2p  = (half_t*)alloc((size_t)4 * 16384 * 2);
  half_t* mm1p  = (half_t*)alloc((size_t)4 * 16384 * 2);
  half_t* mm2p  = (half_t*)alloc((size_t)4 * 16384 * 2);
  half_t* up1p  = (half_t*)alloc((size_t)4 * 32768 * 2);
  half_t* up2p  = (half_t*)alloc((size_t)4 * 16384 * 2);
  half_t* wihp  = (half_t*)alloc((size_t)49152 * 2);
  half_t* whhp  = (half_t*)alloc((size_t)49152 * 2);

  auto pack = [&](const float* W, half_t* o, int K, int Ncols, int tr) {
    int total = K * Ncols;
    pack_w<<<(total + 255) / 256, 256, 0, stream>>>(W, o, K, Ncols, tr);
  };
  pack(enc_w1, enc1p, 128, 128, 0);
  pack(enc_w2, enc2p, 128, 128, 0);
  for (int l = 0; l < 4; ++l) {
    pack(et_w1 + (size_t)l * 36864,  et1p + (size_t)l * 36864, 288, 128, 0);
    pack(et_w2 + (size_t)l * 16384,  et2p + (size_t)l * 16384, 128, 128, 0);
    pack(msg_w1 + (size_t)l * 16384, mm1p + (size_t)l * 16384, 128, 128, 0);
    pack(msg_w2 + (size_t)l * 16384, mm2p + (size_t)l * 16384, 128, 128, 0);
    pack(upd_w1 + (size_t)l * 32768, up1p + (size_t)l * 32768, 256, 128, 0);
    pack(upd_w2 + (size_t)l * 16384, up2p + (size_t)l * 16384, 128, 128, 0);
  }
  pack(gru_wih, wihp, 128, 384, 1);   // W^T: B[k][n] = wih[n][k]
  pack(gru_whh, whhp, 128, 384, 1);

  const int nodeTiles = Nn / 16;   // 3125
  const int edgeTiles = Ee / 32;   // 25000 (32 edges per wave)

  enc_kernel<<<(nodeTiles + WPB - 1) / WPB, 32 * WPB, 0, stream>>>(
      x, enc1p, enc_b1, enc2p, enc_b2, h, h16, nodeTiles);

  for (int l = 0; l < 4; ++l) {
    hipMemsetAsync(aggr, 0, (size_t)Nn * 128 * 4, stream);
    edge_kernel<<<(edgeTiles + EWPB - 1) / EWPB, 32 * EWPB, 0, stream>>>(
        h16, edge_attr, srcI, dstI,
        et1p + (size_t)l * 36864, et_b1 + l * 128,
        et2p + (size_t)l * 16384, et_b2 + l * 128,
        mm1p + (size_t)l * 16384, msg_b1 + l * 128,
        mm2p + (size_t)l * 16384, msg_b2 + l * 128,
        aggr, edgeTiles);
    upd_kernel<<<(nodeTiles + WPB - 1) / WPB, 32 * WPB, 0, stream>>>(
        h16, aggr, h,
        up1p + (size_t)l * 32768, upd_b1 + l * 128,
        up2p + (size_t)l * 16384, upd_b2 + l * 128,
        preln, nodeTiles);
    ln_kernel<<<(Nn * 32 + 255) / 256, 256, 0, stream>>>(preln, ln_g, ln_b, h, h16, Nn);
  }

  gru_kernel<<<(nodeTiles * 8 + WPB - 1) / WPB, 32 * WPB, 0, stream>>>(
      h16, h_prev, wihp, whhp, gru_bih, gru_bhh, (float*)d_out, nodeTiles);
}
